// IPMPDecoder_32487132627123
// MI455X (gfx1250) — compile-verified, matrix operand order checked by reading, not run on hardware
//
#include <hip/hip_runtime.h>
#include <hip/hip_bf16.h>

// Problem constants (from reference)
#define B_    8
#define L_    512
#define K_    30
#define H_    128
#define NIN_  384
#define NPTS_ 8
#define DIN_  584            // H + NUM_IN + 9*NPTS
#define DINP_ 608            // padded to 19*32 for f16 WMMA K-steps
#define KSTEP1 (DINP_/32)    // 19
#define KSTEP2 (H_/32)       // 4
#define FFH_  512            // 4*H
#define EPS_  1e-8f

typedef _Float16 v16h __attribute__((ext_vector_type(16)));
typedef _Float16 v8h  __attribute__((ext_vector_type(8)));
typedef _Float16 v4h  __attribute__((ext_vector_type(4)));
typedef float    v8f  __attribute__((ext_vector_type(8)));

// -------- kernel1 shared memory layout (bytes), phases alias dead regions --------
#define MSG_OFF    0                    // f16 msg[32*608] (through GEMM1); then f32 h3[32*128]
#define H1_OFF     38912                // f16 h1[32*128]
#define H2_OFF     47104                // f16 h2[32*128]
#define HV_OFF     55296                // f32[128]
#define R_OFF      55808                // f32[30*9]
#define T_OFF      56888                // f32[30*3]
#define PLN_OFF    57248                // f32[30*24]
#define PG_OFF     60128                // f32[24]
#define PLOCN_OFF  60224                // f32[8]
#define XBUF_OFF   60256                // f32[128]
#define RED_OFF    60768                // f32[128]
#define SMEM1_BYTES 61280

// -------- kernel2 (batched FFN) shared memory layout --------
#define HRES_OFF   0                    // f32[32*128] residual input; updated in place with FFN out
#define AF16_OFF   16384                // f16[32*128] GEMM-Wi A operand
#define HID_OFF    24576                // f16[32*512] hidden activations
#define SMEM2_BYTES 57344

// -------- workspace layout (f16 halves unless noted) --------
#define WS_W1T  0                                // 128*608
#define WS_W2T  (WS_W1T + H_*DINP_)              // 128*128
#define WS_W3T  (WS_W2T + H_*H_)                 // 128*128
#define WS_WPT  (WS_W3T + H_*H_)                 // 32*128 (24 valid N, padded)
#define WS_WIT  (WS_WPT + 32*H_)                 // 512*128
#define WS_WOT  (WS_WIT + FFH_*H_)               // 128*512
#define WS_HALVES (WS_WOT + H_*FFH_)             // = 245760 halves = 491520 B (16B aligned)

__device__ __forceinline__ float gelu_f(float x) {
    return 0.5f * x * (1.0f + erff(x * 0.7071067811865476f));
}

// LDS byte address of a generic pointer into shared memory (low 32 bits = LDS offset)
__device__ __forceinline__ unsigned int lds_addr_of(const void* p) {
    return (unsigned int)(uintptr_t)p;
}

// Async 16B global->LDS copy (CDNA5 ASYNCcnt domain); pair with async_wait() + barrier.
__device__ __forceinline__ void async_copy_b128(void* lds_dst, const void* gsrc) {
    unsigned int       lo = lds_addr_of(lds_dst);
    unsigned long long ga = (unsigned long long)(uintptr_t)gsrc;
    asm volatile("global_load_async_to_lds_b128 %0, %1, off"
                 :: "v"(lo), "v"(ga) : "memory");
}
__device__ __forceinline__ void async_wait() {
    asm volatile("s_wait_asynccnt 0" ::: "memory");
}

// Load a 16-half WMMA fragment: two contiguous 16B chunks (K=kc..kc+7, kc+16..kc+23)
__device__ __forceinline__ v16h load_frag(const _Float16* p) {
    v8h lo = *(const v8h*)(p);
    v8h hi = *(const v8h*)(p + 16);
    return __builtin_shufflevector(lo, hi, 0,1,2,3,4,5,6,7,8,9,10,11,12,13,14,15);
}

// One wave computes two 16x16 tiles (M-tiles 0 and 1) of a fixed 16-wide N strip.
// A from LDS ([32][lda] f16 row-major), B from global transposed weights
// ([N][ldb] f16, i.e. B[k][n] stored at Bt[n*ldb + k]).
__device__ __forceinline__ void gemm_2x1(const _Float16* Alds, int lda,
                                         const _Float16* Bglb, int ldb,
                                         int ksteps, int nt, int ln, int hi,
                                         v8f& c0, v8f& c1) {
    v8f z = {0.f,0.f,0.f,0.f,0.f,0.f,0.f,0.f};
    c0 = z; c1 = z;
    for (int ks = 0; ks < ksteps; ++ks) {
        const int kc = ks * 32 + hi * 8;
        v16h b  = load_frag(Bglb + (size_t)(nt * 16 + ln) * ldb + kc);
        v16h a0 = load_frag(Alds + (size_t)ln * lda + kc);
        v16h a1 = load_frag(Alds + (size_t)(16 + ln) * lda + kc);
        c0 = __builtin_amdgcn_wmma_f32_16x16x32_f16(false, a0, false, b, (short)0, c0, false, false);
        c1 = __builtin_amdgcn_wmma_f32_16x16x32_f16(false, a1, false, b, (short)0, c1, false, false);
    }
}

// sum-reduce 128 values (threads 0..127 contribute), all 256 threads hit barriers
__device__ __forceinline__ float block_reduce128(float v, float* red, int tid) {
    if (tid < 128) red[tid] = v;
    __syncthreads();
    for (int s = 64; s > 0; s >>= 1) {
        if (tid < s) red[tid] += red[tid + s];
        __syncthreads();
    }
    float r = red[0];
    __syncthreads();
    return r;
}

// Convert all weight matrices to transposed ([N][K]) f16, K-padded where needed.
__global__ void prep_weights_kernel(const float* __restrict__ W1,
                                    const float* __restrict__ W2,
                                    const float* __restrict__ W3,
                                    const float* __restrict__ Wp,
                                    const float* __restrict__ Wi,
                                    const float* __restrict__ Wo,
                                    _Float16* __restrict__ W1t,
                                    _Float16* __restrict__ W2t,
                                    _Float16* __restrict__ W3t,
                                    _Float16* __restrict__ Wpt,
                                    _Float16* __restrict__ Wit,
                                    _Float16* __restrict__ Wot) {
    int i = blockIdx.x * blockDim.x + threadIdx.x;
    if (i < H_ * DINP_) {                       // W1t[n][kk], kk padded 584->608
        int n = i / DINP_, kk = i - n * DINP_;
        W1t[i] = (kk < DIN_) ? (_Float16)W1[kk * H_ + n] : (_Float16)0.0f;
    }
    if (i < H_ * H_) {                          // W2t/W3t[n][kk]
        int n = i / H_, kk = i - n * H_;
        W2t[i] = (_Float16)W2[kk * H_ + n];
        W3t[i] = (_Float16)W3[kk * H_ + n];
    }
    if (i < 32 * H_) {                          // Wpt[n][kk], n padded 24->32
        int n = i >> 7, kk = i & 127;
        Wpt[i] = (n < 24) ? (_Float16)Wp[kk * 24 + n] : (_Float16)0.0f;
    }
    if (i < FFH_ * H_) {                        // Wit[n][kk], n<512, kk<128
        int n = i >> 7, kk = i & 127;
        Wit[i] = (_Float16)Wi[kk * FFH_ + n];
    }
    if (i < H_ * FFH_) {                        // Wot[n][kk], n<128, kk<512
        int n = i / FFH_, kk = i - n * FFH_;
        Wot[i] = (_Float16)Wo[kk * H_ + n];
    }
}

// ---------------- kernel 1: per-node geometry + 3 edge GEMMs + mean + LN1 ----------------
__global__ __launch_bounds__(256)
void ipmp_msg_kernel(
    const float* __restrict__ h_V, const float* __restrict__ h_E,
    const float* __restrict__ Xn,  const float* __restrict__ maskA,
    const float* __restrict__ bp,
    const float* __restrict__ b1,  const float* __restrict__ b2,
    const float* __restrict__ b3,
    const float* __restrict__ g1,  const float* __restrict__ be1,
    const _Float16* __restrict__ W1t, const _Float16* __restrict__ W2t,
    const _Float16* __restrict__ W3t, const _Float16* __restrict__ Wpt,
    float* __restrict__ hlnG)
{
    extern __shared__ __align__(16) char smem[];
    _Float16* msg    = (_Float16*)(smem + MSG_OFF);
    float*    h3     = (float*)   (smem + MSG_OFF);     // aliases msg (dead after GEMM1)
    _Float16* h1     = (_Float16*)(smem + H1_OFF);
    _Float16* h2     = (_Float16*)(smem + H2_OFF);
    float*    hv     = (float*)   (smem + HV_OFF);
    float*    Rb     = (float*)   (smem + R_OFF);
    float*    tb     = (float*)   (smem + T_OFF);
    float*    plnb   = (float*)   (smem + PLN_OFF);
    float*    pglob  = (float*)   (smem + PG_OFF);
    float*    plocn  = (float*)   (smem + PLOCN_OFF);
    float*    xbuf   = (float*)   (smem + XBUF_OFF);
    float*    red    = (float*)   (smem + RED_OFF);

    const int tid  = threadIdx.x;
    const int node = blockIdx.x;            // b*L + l
    const int lane = tid & 31;
    const int ln   = lane & 15;
    const int hi   = (lane >> 4) & 1;
    const int nt   = tid >> 5;              // wave id == N-tile (0..7)

    // ---- phase 0A: async-copy h_V row into LDS, zero msg pads, prefetch W1t ----
    if (tid < 32)
        async_copy_b128(hv + tid * 4, h_V + (size_t)node * H_ + tid * 4);
    for (int i = tid; i < 32 * DINP_; i += 256) msg[i] = (_Float16)0.0f;
    for (int i = tid; i < (H_ * DINP_ * 2) / 128; i += 256)      // warm W1t into GL2
        __builtin_prefetch((const char*)W1t + (size_t)i * 128, 0, 1);
    async_wait();
    __syncthreads();

    // ---- phase 0B: stream h_E once (vectorized b128 -> packed f16), frames ----
    for (int i = tid; i < K_ * (NIN_ / 4); i += 256) {      // 30 * 96 float4
        int k = i / 96, c4 = i - k * 96;
        float4 e = ((const float4*)(h_E + ((size_t)node * K_ + k) * NIN_))[c4];
        v4h h4 = {(_Float16)e.x, (_Float16)e.y, (_Float16)e.z, (_Float16)e.w};
        *(v4h*)(msg + k * DINP_ + H_ + c4 * 4) = h4;
    }
    if (tid < K_) {
        const float* x = Xn + ((size_t)node * K_ + tid) * 9;
        float Nx=x[0],Ny=x[1],Nz=x[2], Ax=x[3],Ay=x[4],Az=x[5], Cx=x[6],Cy=x[7],Cz=x[8];
        float e0x=Ax-Nx, e0y=Ay-Ny, e0z=Az-Nz;
        float inv0 = 1.0f / sqrtf(e0x*e0x + e0y*e0y + e0z*e0z + EPS_);
        e0x*=inv0; e0y*=inv0; e0z*=inv0;
        float e1x=Cx-Ax, e1y=Cy-Ay, e1z=Cz-Az;
        float pr = e0x*e1x + e0y*e1y + e0z*e1z;
        e1x-=e0x*pr; e1y-=e0y*pr; e1z-=e0z*pr;
        float inv1 = 1.0f / sqrtf(e1x*e1x + e1y*e1y + e1z*e1z + EPS_);
        e1x*=inv1; e1y*=inv1; e1z*=inv1;
        float e2x=e0y*e1z-e0z*e1y, e2y=e0z*e1x-e0x*e1z, e2z=e0x*e1y-e0y*e1x;
        float* Rk = Rb + tid*9;                 // R[i][j], columns = e0,e1,e2
        Rk[0]=e0x; Rk[1]=e1x; Rk[2]=e2x;
        Rk[3]=e0y; Rk[4]=e1y; Rk[5]=e2y;
        Rk[6]=e0z; Rk[7]=e1z; Rk[8]=e2z;
        float* tk = tb + tid*3; tk[0]=Ax; tk[1]=Ay; tk[2]=Az;
    }
    __syncthreads();

    // ---- phase 0C: p_ln via WMMA (waves 0-1): A = msg cols 384..511 (h_V_prev f16) ----
    if (nt < 2) {
        v8f c0, c1;
        gemm_2x1(msg + 384, DINP_, Wpt, H_, KSTEP2, nt, ln, hi, c0, c1);
        int n = nt * 16 + ln;
        if (n < 24) {
            float bias = bp[n];
            for (int r = 0; r < 8; ++r) {
                int m0 = r + hi * 8;                     // < 16 < K
                plnb[m0 * 24 + n] = c0[r] + bias;
                int m1 = 16 + m0;
                if (m1 < K_) plnb[m1 * 24 + n] = c1[r] + bias;
            }
        }
    }
    // node_expand: msg cols 0..127 = h_V (vectorized)
    for (int i = tid; i < K_ * (H_ / 4); i += 256) {        // 30 * 32 float4
        int k = i >> 5, c4 = i & 31;
        float4 v = ((const float4*)hv)[c4];
        v4h h4 = {(_Float16)v.x, (_Float16)v.y, (_Float16)v.z, (_Float16)v.w};
        *(v4h*)(msg + k * DINP_ + c4 * 4) = h4;
    }
    __syncthreads();

    // ---- phase 0D: p_global (k=0), ||p_local|| ----
    if (tid < 24) {
        int n = tid / 3, i2 = tid - n * 3;
        const float* p0 = plnb + n * 3;                     // k = 0 row
        pglob[tid] = Rb[i2*3+0]*p0[0] + Rb[i2*3+1]*p0[1] + Rb[i2*3+2]*p0[2] + tb[i2];
    }
    if (tid >= 32 && tid < 40) {
        int n = tid - 32;
        const float* p0 = plnb + n * 3;
        plocn[n] = sqrtf(p0[0]*p0[0] + p0[1]*p0[1] + p0[2]*p0[2] + EPS_);
    }
    __syncthreads();

    // ---- phase 0E: geometric features -> msg cols 512..583 ----
    for (int task = tid; task < K_ * NPTS_; task += 256) {
        int k = task >> 3, n = task & 7;
        const float* Rk = Rb + k * 9;
        const float* tk = tb + k * 3;
        const float* p  = plnb + k * 24 + n * 3;
        float gx = Rk[0]*p[0] + Rk[1]*p[1] + Rk[2]*p[2] + tk[0];
        float gy = Rk[3]*p[0] + Rk[4]*p[1] + Rk[5]*p[2] + tk[1];
        float gz = Rk[6]*p[0] + Rk[7]*p[1] + Rk[8]*p[2] + tk[2];
        float d0 = gx - tb[0], d1 = gy - tb[1], d2 = gz - tb[2];
        float l0 = Rb[0]*d0 + Rb[3]*d1 + Rb[6]*d2;          // npl = R0^T * d
        float l1 = Rb[1]*d0 + Rb[4]*d1 + Rb[7]*d2;
        float l2 = Rb[2]*d0 + Rb[5]*d1 + Rb[8]*d2;
        float nln = sqrtf(l0*l0 + l1*l1 + l2*l2 + EPS_);
        float q0 = pglob[n*3+0]-gx, q1 = pglob[n*3+1]-gy, q2 = pglob[n*3+2]-gz;
        float ngn = sqrtf(q0*q0 + q1*q1 + q2*q2 + EPS_);
        _Float16* row = msg + k * DINP_;
        const float* pl = plnb + n * 3;                     // p_local (k=0), broadcast over k
        row[512 + n*3 + 0] = (_Float16)pl[0];
        row[512 + n*3 + 1] = (_Float16)pl[1];
        row[512 + n*3 + 2] = (_Float16)pl[2];
        row[536 + n]       = (_Float16)plocn[n];
        row[544 + n*3 + 0] = (_Float16)l0;
        row[544 + n*3 + 1] = (_Float16)l1;
        row[544 + n*3 + 2] = (_Float16)l2;
        row[568 + n]       = (_Float16)nln;
        row[576 + n]       = (_Float16)ngn;
    }
    __syncthreads();

    // ---- GEMM1: msg(32x608) @ W1 -> gelu -> h1 (f16) ----
    v8f c0, c1;
    gemm_2x1(msg, DINP_, W1t, DINP_, KSTEP1, nt, ln, hi, c0, c1);
    {
        int n = nt*16 + ln;
        float bias = b1[n];
        for (int r = 0; r < 8; ++r) {
            int m0 = r + hi*8;
            h1[m0 * H_ + n]        = (_Float16)gelu_f(c0[r] + bias);
            h1[(16 + m0) * H_ + n] = (_Float16)gelu_f(c1[r] + bias);
        }
    }
    __syncthreads();

    // ---- GEMM2: h1 @ W2 -> gelu -> h2 (f16) ----
    gemm_2x1(h1, H_, W2t, H_, KSTEP2, nt, ln, hi, c0, c1);
    {
        int n = nt*16 + ln;
        float bias = b2[n];
        for (int r = 0; r < 8; ++r) {
            int m0 = r + hi*8;
            h2[m0 * H_ + n]        = (_Float16)gelu_f(c0[r] + bias);
            h2[(16 + m0) * H_ + n] = (_Float16)gelu_f(c1[r] + bias);
        }
    }
    __syncthreads();

    // ---- GEMM3: h2 @ W3 + b3, masked -> h3 (f32, aliases msg) ----
    gemm_2x1(h2, H_, W3t, H_, KSTEP2, nt, ln, hi, c0, c1);
    {
        int n = nt*16 + ln;
        float bias = b3[n];
        for (int r = 0; r < 8; ++r) {
            int m0 = r + hi*8;
            int m1 = 16 + m0;
            float mk0 = maskA[(size_t)node * K_ + m0];
            float mk1 = (m1 < K_) ? maskA[(size_t)node * K_ + m1] : 0.0f;
            h3[m0 * H_ + n] = (c0[r] + bias) * mk0;
            h3[m1 * H_ + n] = (c1[r] + bias) * mk1;
        }
    }
    __syncthreads();

    // ---- mean over K, residual, LN1, stage to global ----
    if (tid < H_) {
        float s = 0.f;
        for (int k = 0; k < K_; ++k) s += h3[k * H_ + tid];
        xbuf[tid] = hv[tid] + s * (1.0f / 30.0f);
    }
    __syncthreads();
    {
        float x  = (tid < H_) ? xbuf[tid] : 0.f;
        float mu = block_reduce128(x, red, tid) * (1.0f / 128.0f);
        float dc = (tid < H_) ? (x - mu) : 0.f;
        float vr = block_reduce128(dc * dc, red, tid) * (1.0f / 128.0f);
        if (tid < H_)
            hlnG[(size_t)node * H_ + tid] =
                dc * (1.0f / sqrtf(vr + 1e-5f)) * g1[tid] + be1[tid];
    }
}

// ---------------- kernel 2: batched FFN + LN2 over all 4096 nodes ----------------
__global__ __launch_bounds__(256)
void ipmp_ffn_kernel(
    const float* __restrict__ hlnG, const float* __restrict__ maskV,
    const float* __restrict__ bi,   const float* __restrict__ bo,
    const float* __restrict__ g2,   const float* __restrict__ be2,
    const _Float16* __restrict__ Wit, const _Float16* __restrict__ Wot,
    float* __restrict__ out)
{
    extern __shared__ __align__(16) char smem[];
    float*    hres = (float*)   (smem + HRES_OFF);   // f32 residual, updated in place
    _Float16* af16 = (_Float16*)(smem + AF16_OFF);   // A operand
    _Float16* hid  = (_Float16*)(smem + HID_OFF);    // hidden (32x512)

    const int tid  = threadIdx.x;
    const int base = blockIdx.x * 32;               // first node row of this tile
    const int lane = tid & 31;
    const int ln   = lane & 15;
    const int hi   = (lane >> 4) & 1;
    const int wv   = tid >> 5;

    // async-copy the 32x128 f32 tile of LN1 output straight into LDS (raw bytes),
    // prefetch FFN weights into GL2 meanwhile
    for (int i = tid; i < 1024; i += 256)           // 1024 x 16B = 16 KB
        async_copy_b128(hres + i * 4, hlnG + (size_t)base * H_ + i * 4);
    for (int i = tid; i < 1024; i += 256)           // warm Wit (128 KB)
        __builtin_prefetch((const char*)Wit + (size_t)i * 128, 0, 1);
    for (int i = tid; i < 1024; i += 256)           // warm Wot (128 KB)
        __builtin_prefetch((const char*)Wot + (size_t)i * 128, 0, 1);
    async_wait();
    __syncthreads();

    // f16 copy of the tile for the GEMM-Wi A operand
    for (int i = tid; i < 1024; i += 256) {
        float4 v = ((const float4*)hres)[i];
        v4h h4 = {(_Float16)v.x, (_Float16)v.y, (_Float16)v.z, (_Float16)v.w};
        *(v4h*)(af16 + i * 4) = h4;
    }
    __syncthreads();

    // GEMM-Wi: (32x128) @ (128x512) -> gelu -> hid ; each wave owns 4 N-tiles
    for (int j = 0; j < 4; ++j) {
        int nt = wv * 4 + j;
        v8f c0, c1;
        gemm_2x1(af16, H_, Wit, H_, KSTEP2, nt, ln, hi, c0, c1);
        int n = nt * 16 + ln;
        float bias = bi[n];
        for (int r = 0; r < 8; ++r) {
            int m0 = r + hi * 8;
            hid[m0 * FFH_ + n]        = (_Float16)gelu_f(c0[r] + bias);
            hid[(16 + m0) * FFH_ + n] = (_Float16)gelu_f(c1[r] + bias);
        }
    }
    __syncthreads();

    // GEMM-Wo: (32x512) @ (512x128) + bo + residual (in place into hres)
    {
        v8f c0, c1;
        gemm_2x1(hid, FFH_, Wot, FFH_, FFH_ / 32, wv, ln, hi, c0, c1);
        int n = wv * 16 + ln;
        float bias = bo[n];
        for (int r = 0; r < 8; ++r) {
            int m0 = r + hi * 8;
            int m1 = 16 + m0;
            hres[m0 * H_ + n] += c0[r] + bias;
            hres[m1 * H_ + n] += c1[r] + bias;
        }
    }
    __syncthreads();

    // LN2 + mask_V + store: one thread per row
    if (tid < 32) {
        const float* row = hres + tid * H_;
        float s = 0.f;
        for (int c = 0; c < H_; ++c) s += row[c];
        float mu = s * (1.0f / 128.0f);
        float v = 0.f;
        for (int c = 0; c < H_; ++c) { float d = row[c] - mu; v += d * d; }
        float inv = 1.0f / sqrtf(v * (1.0f / 128.0f) + 1e-5f);
        int node = base + tid;
        float mv = maskV[node];
        float* o = out + (size_t)node * H_;
        for (int c = 0; c < H_; ++c)
            o[c] = mv * ((row[c] - mu) * inv * g2[c] + be2[c]);
    }
}

extern "C" void kernel_launch(void* const* d_in, const int* in_sizes, int n_in,
                              void* d_out, int out_size, void* d_ws, size_t ws_size,
                              hipStream_t stream) {
    (void)in_sizes; (void)n_in; (void)out_size; (void)ws_size;
    const float* h_V    = (const float*)d_in[0];
    const float* h_E    = (const float*)d_in[1];
    /* d_in[2] = E_idx (unused by the forward pass) */
    const float* Xn     = (const float*)d_in[3];
    const float* maskV  = (const float*)d_in[4];
    const float* maskA  = (const float*)d_in[5];
    const float* Wp     = (const float*)d_in[6];
    const float* bp     = (const float*)d_in[7];
    const float* W1     = (const float*)d_in[8];
    const float* b1     = (const float*)d_in[9];
    const float* W2     = (const float*)d_in[10];
    const float* b2     = (const float*)d_in[11];
    const float* W3     = (const float*)d_in[12];
    const float* b3     = (const float*)d_in[13];
    const float* Wi     = (const float*)d_in[14];
    const float* bi     = (const float*)d_in[15];
    const float* Wo     = (const float*)d_in[16];
    const float* bo     = (const float*)d_in[17];
    const float* g1     = (const float*)d_in[18];
    const float* be1    = (const float*)d_in[19];
    const float* g2     = (const float*)d_in[20];
    const float* be2    = (const float*)d_in[21];

    _Float16* wsh = (_Float16*)d_ws;
    _Float16* W1t = wsh + WS_W1T;
    _Float16* W2t = wsh + WS_W2T;
    _Float16* W3t = wsh + WS_W3T;
    _Float16* Wpt = wsh + WS_WPT;
    _Float16* Wit = wsh + WS_WIT;
    _Float16* Wot = wsh + WS_WOT;
    float*   hlnG = (float*)((char*)d_ws + WS_HALVES * sizeof(_Float16));  // 4096*128 f32

    prep_weights_kernel<<<(H_ * DINP_ + 255) / 256, 256, 0, stream>>>(
        W1, W2, W3, Wp, Wi, Wo, W1t, W2t, W3t, Wpt, Wit, Wot);

    ipmp_msg_kernel<<<B_ * L_, 256, SMEM1_BYTES, stream>>>(
        h_V, h_E, Xn, maskA, bp, b1, b2, b3, g1, be1,
        W1t, W2t, W3t, Wpt, hlnG);

    ipmp_ffn_kernel<<<(B_ * L_) / 32, 256, SMEM2_BYTES, stream>>>(
        hlnG, maskV, bi, bo, g2, be2, Wit, Wot, (float*)d_out);
}
